// GNNModel_50319836840080
// MI455X (gfx1250) — matrix operand order repaired and checked by reading, not compiled, not run on hardware
//
#include <hip/hip_runtime.h>
#include <hip/hip_bf16.h>

#define D 128
#define TWO_D 256
#define LAYERS 4
#define EPB 64          // edges per block in edge kernel
#define NEG_INF_BITS 0xFF800000u

typedef __attribute__((ext_vector_type(16))) __bf16 v16bf;
typedef __attribute__((ext_vector_type(8)))  __bf16 bf16x8;
typedef __attribute__((ext_vector_type(8)))  float  v8f;

// ---------------- helpers ----------------

__device__ __forceinline__ float geluf(float x) {
    return 0.5f * x * (1.0f + erff(x * 0.70710678118654752f));
}

// float atomic max via monotone integer encodings
__device__ __forceinline__ void atomicMaxF(float* addr, float val) {
    if (!(val < 0.0f)) atomicMax((int*)addr, __float_as_int(val));
    else               atomicMin((unsigned int*)addr, __float_as_uint(val));
}

// row LayerNorm (no gamma/beta) across a 128-thread block; returns (v-mean)*rsqrt(var+eps)
__device__ __forceinline__ float block_ln_128(float v, float* red, float* stat) {
    const int tid = threadIdx.x;
    red[tid] = v; __syncthreads();
    #pragma unroll
    for (int s = 64; s > 0; s >>= 1) { if (tid < s) red[tid] += red[tid + s]; __syncthreads(); }
    if (tid == 0) stat[0] = red[0] * (1.0f / 128.0f);
    __syncthreads();
    const float mean = stat[0];
    const float d = v - mean;
    red[tid] = d * d; __syncthreads();
    #pragma unroll
    for (int s = 64; s > 0; s >>= 1) { if (tid < s) red[tid] += red[tid + s]; __syncthreads(); }
    if (tid == 0) stat[1] = rsqrtf(red[0] * (1.0f / 128.0f) + 1e-5f);
    __syncthreads();
    const float r = stat[1];
    __syncthreads();
    return d * r;
}

// load a 16x32 bf16 A-fragment's per-lane data: two contiguous 8-element chunks
__device__ __forceinline__ v16bf load_frag16(const __bf16* base) {
    bf16x8 lo = *(const bf16x8*)(base);
    bf16x8 hi = *(const bf16x8*)(base + 16);
    return __builtin_shufflevector(lo, hi, 0,1,2,3,4,5,6,7,8,9,10,11,12,13,14,15);
}

// ---------------- small utility kernels ----------------

__global__ void k_fill_u32(unsigned int* p, unsigned int val, int n) {
    int i = blockIdx.x * 256 + threadIdx.x;
    if (i < n) p[i] = val;
}

// Pre-shuffle a [K,128] f32 weight matrix into WMMA-fragment-major bf16 layout:
// out[((nt*KB + kb)*32 + lane)*16 + j] = W[(kb*32 + 8*(lane>>4) + (j&7) + 16*(j>>3))*128 + nt*16 + (lane&15)]
// so each lane's 16 B-matrix values are 32 contiguous bytes (2x global_load_b128).
__global__ void k_shuffleW(const float* __restrict__ W, __bf16* __restrict__ out,
                           int KB, int n) {
    int i = blockIdx.x * 256 + threadIdx.x;
    if (i >= n) return;
    const int j    = i & 15;
    const int lane = (i >> 4) & 31;
    const int fi   = i >> 9;           // fragment index = nt*KB + kb
    const int kb   = fi % KB;
    const int nt   = fi / KB;
    const int k    = kb * 32 + ((lane >> 4) << 3) + (j & 7) + ((j >> 3) << 4);
    const int col  = nt * 16 + (lane & 15);
    out[i] = (__bf16)W[k * 128 + col];
}

__global__ void k_degree(const int* __restrict__ dst, float* deg, int E) {
    int i = blockIdx.x * 256 + threadIdx.x;
    if (i < E) unsafeAtomicAdd(&deg[dst[i]], 1.0f);
}

// ---------------- input net: Linear(4->128) LN GELU Linear(128->128) LN ----------------

__global__ void k_input_net(const float* __restrict__ x,
                            const float* __restrict__ W1, const float* __restrict__ b1,
                            const float* __restrict__ g1, const float* __restrict__ bn1,
                            const float* __restrict__ W2, const float* __restrict__ b2,
                            const float* __restrict__ g2, const float* __restrict__ bn2,
                            float* __restrict__ xcur) {
    const int n = blockIdx.x, c = threadIdx.x;
    __shared__ float red[128]; __shared__ float stat[2]; __shared__ float s1[128];
    const float x0 = x[n*4+0], x1 = x[n*4+1], x2 = x[n*4+2], x3 = x[n*4+3];
    float t = b1[c] + x0*W1[c] + x1*W1[128+c] + x2*W1[256+c] + x3*W1[384+c];
    float ln = block_ln_128(t, red, stat);
    s1[c] = geluf(ln * g1[c] + bn1[c]);
    __syncthreads();
    float t2 = b2[c];
    #pragma unroll 4
    for (int k = 0; k < 128; ++k) t2 += s1[k] * W2[k*128+c];
    float ln2 = block_ln_128(t2, red, stat);
    xcur[n*128+c] = ln2 * g2[c] + bn2[c];
}

// segment max/sum/count over batch for a [N,128] feature matrix
__global__ void k_seg_graph(const float* __restrict__ h, const int* __restrict__ batch,
                            float* gmax, float* gsum, float* gcnt, int do_count) {
    const int n = blockIdx.x, c = threadIdx.x;
    const int g = batch[n];
    const float v = h[n*128+c];
    atomicMaxF(&gmax[g*128+c], v);
    unsafeAtomicAdd(&gsum[g*128+c], v);
    if (do_count && c == 0) unsafeAtomicAdd(&gcnt[g], 1.0f);
}

// pre-LN per layer, emits bf16 node features for the edge gather
__global__ void k_preln(const float* __restrict__ xcur,
                        const float* __restrict__ g, const float* __restrict__ b,
                        __bf16* __restrict__ hnb) {
    const int n = blockIdx.x, c = threadIdx.x;
    __shared__ float red[128]; __shared__ float stat[2];
    float ln = block_ln_128(xcur[n*128+c], red, stat);
    hnb[n*128+c] = (__bf16)(ln * g[c] + b[c]);
}

__global__ void k_init_aggr(float* amax, float* asum, int n) {
    int i = blockIdx.x * 256 + threadIdx.x;
    if (i < n) { ((unsigned int*)amax)[i] = NEG_INF_BITS; asum[i] = 0.0f; }
}

// ---------------- the WMMA edge-MLP kernel ----------------
// 64 edges per block, 128 threads (4 waves). Per wave: 2 column tiles of 16.
// GEMM1: [64,256]x[256,128] bf16 -> 64 v_wmma/wave; GEMM2: [64,128]x[128,128] -> 32.

__global__ void __launch_bounds__(128)
k_edge_mlp(const __bf16* __restrict__ hnb,
           const int* __restrict__ src, const int* __restrict__ dst,
           const __bf16* __restrict__ W1s, const float* __restrict__ cb1,
           const float* __restrict__ cg1, const float* __restrict__ cbb1,
           const __bf16* __restrict__ W2s, const float* __restrict__ cb2,
           const float* __restrict__ cg2, const float* __restrict__ cbb2,
           float* __restrict__ amax, float* __restrict__ asum, int E) {
    const int tid  = threadIdx.x;
    const int lane = tid & 31;
    const int wid  = tid >> 5;
    const int e0   = blockIdx.x * EPB;

    __shared__ float  s_out[EPB * D];       // 32KB, f32 GEMM output; aliased as bf16 m matrix
    __shared__ __bf16 s_h2[EPB * D];        // 16KB, bf16 activations for GEMM2
    __bf16* s_m1 = reinterpret_cast<__bf16*>(s_out);   // [64][256] bf16

    // ---- gather: m = [xi, xj - xi] into LDS (bf16), 16B vector moves ----
    {
        const int r    = tid >> 1;      // 64 rows, 2 threads per row
        const int half = tid & 1;       // 8 chunks of 8 cols each
        const int e    = e0 + r;
        int dn = 0, sn = 0, valid = 0;
        if (e < E) { dn = dst[e]; sn = src[e]; valid = 1; }
        #pragma unroll
        for (int ch8 = 0; ch8 < 8; ++ch8) {
            const int ch = (half * 8 + ch8) * 8;     // column offset, multiple of 8
            bf16x8 vi, vd;
            if (valid) {
                vi = *(const bf16x8*)(hnb + (size_t)dn * 128 + ch);
                bf16x8 vj = *(const bf16x8*)(hnb + (size_t)sn * 128 + ch);
                #pragma unroll
                for (int j = 0; j < 8; ++j)
                    vd[j] = (__bf16)((float)vj[j] - (float)vi[j]);
            } else {
                #pragma unroll
                for (int j = 0; j < 8; ++j) { vi[j] = (__bf16)0.0f; vd[j] = (__bf16)0.0f; }
            }
            *(bf16x8*)(s_m1 + r * 256 + ch)       = vi;
            *(bf16x8*)(s_m1 + r * 256 + 128 + ch) = vd;
        }
    }
    __syncthreads();

    const int mrow   = lane & 15;
    const int khalfA = (lane >> 4) * 8;     // A-fragment K offset (16x32 bf16 layout)
    const int chalf  = (lane >> 4) * 8;     // C row offset

    // ---- GEMM1: K=256, fragments from pre-shuffled weights ----
    v8f acc[2][4];
    #pragma unroll
    for (int ct = 0; ct < 2; ++ct) {
        const int nt   = wid * 2 + ct;          // column tile 0..7
        const int ncol = nt * 16 + mrow;
        const float bias = cb1[ncol];
        #pragma unroll
        for (int rt = 0; rt < 4; ++rt)
            #pragma unroll
            for (int v = 0; v < 8; ++v) acc[ct][rt][v] = bias;
        for (int kb = 0; kb < 8; ++kb) {
            // per-lane 32B contiguous -> 2x global_load_b128
            v16bf bfrag = *(const v16bf*)(W1s + ((size_t)(nt * 8 + kb) * 32 + lane) * 16);
            const int kA = kb * 32 + khalfA;
            #pragma unroll
            for (int rt = 0; rt < 4; ++rt) {
                v16bf afrag = load_frag16(s_m1 + (rt * 16 + mrow) * 256 + kA);
                acc[ct][rt] = __builtin_amdgcn_wmma_f32_16x16x32_bf16(
                    false, afrag, false, bfrag, (short)0, acc[ct][rt], false, false);
            }
        }
    }
    __syncthreads();   // done reading s_m1; safe to overwrite with f32 output
    #pragma unroll
    for (int ct = 0; ct < 2; ++ct) {
        const int ncol = (wid * 2 + ct) * 16 + mrow;
        #pragma unroll
        for (int rt = 0; rt < 4; ++rt)
            #pragma unroll
            for (int v = 0; v < 8; ++v)
                s_out[(rt*16 + v + chalf)*128 + ncol] = acc[ct][rt][v];
    }
    __syncthreads();

    // ---- LN + GELU rows -> bf16 ----
    if (tid < EPB) {
        float mean = 0.0f;
        for (int c = 0; c < 128; ++c) mean += s_out[tid*128 + c];
        mean *= (1.0f / 128.0f);
        float var = 0.0f;
        for (int c = 0; c < 128; ++c) { float d = s_out[tid*128 + c] - mean; var += d*d; }
        const float r = rsqrtf(var * (1.0f / 128.0f) + 1e-5f);
        for (int c = 0; c < 128; ++c) {
            float v = (s_out[tid*128 + c] - mean) * r * cg1[c] + cbb1[c];
            s_h2[tid*128 + c] = (__bf16)geluf(v);
        }
    }
    __syncthreads();

    // ---- GEMM2: K=128 ----
    #pragma unroll
    for (int ct = 0; ct < 2; ++ct) {
        const int nt   = wid * 2 + ct;
        const int ncol = nt * 16 + mrow;
        const float bias = cb2[ncol];
        #pragma unroll
        for (int rt = 0; rt < 4; ++rt)
            #pragma unroll
            for (int v = 0; v < 8; ++v) acc[ct][rt][v] = bias;
        for (int kb = 0; kb < 4; ++kb) {
            v16bf bfrag = *(const v16bf*)(W2s + ((size_t)(nt * 4 + kb) * 32 + lane) * 16);
            const int kA = kb * 32 + khalfA;
            #pragma unroll
            for (int rt = 0; rt < 4; ++rt) {
                v16bf afrag = load_frag16(s_h2 + (rt * 16 + mrow) * 128 + kA);
                acc[ct][rt] = __builtin_amdgcn_wmma_f32_16x16x32_bf16(
                    false, afrag, false, bfrag, (short)0, acc[ct][rt], false, false);
            }
        }
    }
    __syncthreads();
    #pragma unroll
    for (int ct = 0; ct < 2; ++ct) {
        const int ncol = (wid * 2 + ct) * 16 + mrow;
        #pragma unroll
        for (int rt = 0; rt < 4; ++rt)
            #pragma unroll
            for (int v = 0; v < 8; ++v)
                s_out[(rt*16 + v + chalf)*128 + ncol] = acc[ct][rt][v];
    }
    __syncthreads();

    // ---- LN + GELU rows in place ----
    if (tid < EPB) {
        float mean = 0.0f;
        for (int c = 0; c < 128; ++c) mean += s_out[tid*128 + c];
        mean *= (1.0f / 128.0f);
        float var = 0.0f;
        for (int c = 0; c < 128; ++c) { float d = s_out[tid*128 + c] - mean; var += d*d; }
        const float r = rsqrtf(var * (1.0f / 128.0f) + 1e-5f);
        for (int c = 0; c < 128; ++c) {
            float v = (s_out[tid*128 + c] - mean) * r * cg2[c] + cbb2[c];
            s_out[tid*128 + c] = geluf(v);
        }
    }
    __syncthreads();

    // ---- scatter-aggregate: per-dst max and sum ----
    for (int idx = tid; idx < EPB * D; idx += 128) {
        const int r = idx >> 7, c = idx & 127;
        const int e = e0 + r;
        if (e < E) {
            const int d = dst[e];
            const float v = s_out[r*128 + c];
            atomicMaxF(&amax[d*128 + c], v);
            unsafeAtomicAdd(&asum[d*128 + c], v);
        }
    }
}

// ---------------- node update: xcur += [a_max, a_mean] @ pW + pb ----------------

__global__ void k_node_update(const float* __restrict__ amax, const float* __restrict__ asum,
                              const float* __restrict__ deg,
                              const float* __restrict__ pW, const float* __restrict__ pb,
                              float* __restrict__ xcur) {
    const int n = blockIdx.x, c = threadIdx.x;
    __shared__ float aggr[256];
    float mx = amax[n*128+c];
    if (!(fabsf(mx) < 1e38f)) mx = 0.0f;          // empty segments / NaN -> 0
    const float cnt = fmaxf(deg[n], 1.0f);
    aggr[c]       = mx;
    aggr[128 + c] = asum[n*128+c] / cnt;
    __syncthreads();
    float acc = pb[c];
    #pragma unroll 4
    for (int k = 0; k < 256; ++k) acc += aggr[k] * pW[k*128+c];
    xcur[n*128+c] += acc;
}

// final LN then graph segment max/sum
__global__ void k_final_seg(const float* __restrict__ xcur,
                            const float* __restrict__ ag_g, const float* __restrict__ ag_b,
                            const int* __restrict__ batch,
                            float* g2max, float* g2sum) {
    const int n = blockIdx.x, c = threadIdx.x;
    __shared__ float red[128]; __shared__ float stat[2];
    float ln = block_ln_128(xcur[n*128+c], red, stat);
    const float v = ln * ag_g[c] + ag_b[c];
    const int g = batch[n];
    atomicMaxF(&g2max[g*128+c], v);
    unsafeAtomicAdd(&g2sum[g*128+c], v);
}

// ---------------- graph head ----------------

__global__ void k_graph_head(const float* __restrict__ gmax, const float* __restrict__ gsum,
                             const float* __restrict__ gcnt,
                             const float* __restrict__ g2max, const float* __restrict__ g2sum,
                             const float* kW1, const float* kb1, const float* kg1, const float* kbb1,
                             const float* kW2, const float* kb2, const float* kg2, const float* kbb2,
                             const float* kW3, const float* kb3, const float* kg3, const float* kbb3,
                             const float* kW4, const float* kb4, float* __restrict__ out) {
    const int g = blockIdx.x, c = threadIdx.x;
    __shared__ float xp[512]; __shared__ float z[128];
    __shared__ float red[128]; __shared__ float stat[2]; __shared__ float z3[32];
    const float cnt = fmaxf(gcnt[g], 1.0f);
    float m1v = gmax[g*128+c];  if (!(fabsf(m1v) < 1e38f)) m1v = 0.0f;
    float m2v = g2max[g*128+c]; if (!(fabsf(m2v) < 1e38f)) m2v = 0.0f;
    xp[c]       = m1v;
    xp[128 + c] = gsum[g*128+c] / cnt;
    xp[256 + c] = m2v;
    xp[384 + c] = g2sum[g*128+c] / cnt;
    __syncthreads();
    float a = kb1[c];
    for (int k = 0; k < 512; ++k) a += xp[k] * kW1[k*128+c];
    float ln = block_ln_128(a, red, stat);
    z[c] = geluf(ln * kg1[c] + kbb1[c]);
    __syncthreads();
    float a2 = kb2[c];
    for (int k = 0; k < 128; ++k) a2 += z[k] * kW2[k*128+c];
    ln = block_ln_128(a2, red, stat);
    const float z2v = geluf(ln * kg2[c] + kbb2[c]);
    __syncthreads();
    z[c] = z2v;
    __syncthreads();
    if (c < 32) {
        float a3 = kb3[c];
        for (int k = 0; k < 128; ++k) a3 += z[k] * kW3[k*32+c];
        z3[c] = a3;
    }
    __syncthreads();
    if (c == 0) {
        float mean = 0.0f;
        for (int i = 0; i < 32; ++i) mean += z3[i];
        mean *= (1.0f / 32.0f);
        float var = 0.0f;
        for (int i = 0; i < 32; ++i) { float d = z3[i] - mean; var += d*d; }
        const float r = rsqrtf(var * (1.0f / 32.0f) + 1e-5f);
        float acc = kb4[0];
        for (int i = 0; i < 32; ++i) {
            float v = geluf((z3[i] - mean) * r * kg3[i] + kbb3[i]);
            acc += v * kW4[i];
        }
        out[g] = acc;
    }
}

// ---------------- host side ----------------

static inline size_t alignup(size_t x) { return (x + 255) & ~size_t(255); }

extern "C" void kernel_launch(void* const* d_in, const int* in_sizes, int n_in,
                              void* d_out, int out_size, void* d_ws, size_t ws_size,
                              hipStream_t stream) {
    const int N = in_sizes[2];
    const int E = in_sizes[1] / 2;
    const int G = out_size;            // output is [G,1] f32

    const float* x      = (const float*)d_in[0];
    const int*   ei     = (const int*)d_in[1];
    const int*   srcI   = ei;
    const int*   dstI   = ei + E;
    const int*   batch  = (const int*)d_in[2];
    const float* in_W1  = (const float*)d_in[4];
    const float* in_b1  = (const float*)d_in[5];
    const float* in_g1  = (const float*)d_in[6];
    const float* in_bn1 = (const float*)d_in[7];
    const float* in_W2  = (const float*)d_in[8];
    const float* in_b2  = (const float*)d_in[9];
    const float* in_g2  = (const float*)d_in[10];
    const float* in_bn2 = (const float*)d_in[11];
    const float* pre_g  = (const float*)d_in[12];
    const float* pre_b  = (const float*)d_in[13];
    const float* cW1    = (const float*)d_in[14];
    const float* cb1    = (const float*)d_in[15];
    const float* cg1    = (const float*)d_in[16];
    const float* cbb1   = (const float*)d_in[17];
    const float* cW2    = (const float*)d_in[18];
    const float* cb2    = (const float*)d_in[19];
    const float* cg2    = (const float*)d_in[20];
    const float* cbb2   = (const float*)d_in[21];
    const float* pW     = (const float*)d_in[22];
    const float* pb     = (const float*)d_in[23];
    const float* ag_g   = (const float*)d_in[24];
    const float* ag_b   = (const float*)d_in[25];
    const float* kW1    = (const float*)d_in[26];
    const float* kb1    = (const float*)d_in[27];
    const float* kg1    = (const float*)d_in[28];
    const float* kbb1   = (const float*)d_in[29];
    const float* kW2    = (const float*)d_in[30];
    const float* kb2    = (const float*)d_in[31];
    const float* kg2    = (const float*)d_in[32];
    const float* kbb2   = (const float*)d_in[33];
    const float* kW3    = (const float*)d_in[34];
    const float* kb3    = (const float*)d_in[35];
    const float* kg3    = (const float*)d_in[36];
    const float* kbb3   = (const float*)d_in[37];
    const float* kW4    = (const float*)d_in[38];
    const float* kb4    = (const float*)d_in[39];

    // workspace carve-up
    char* ws = (char*)d_ws;
    size_t off = 0;
    float*  xcur = (float*)(ws + off);  off = alignup(off + (size_t)N * D * 4);
    float*  amax = (float*)(ws + off);  off = alignup(off + (size_t)N * D * 4);
    float*  asum = (float*)(ws + off);  off = alignup(off + (size_t)N * D * 4);
    __bf16* hnb  = (__bf16*)(ws + off); off = alignup(off + (size_t)N * D * 2);
    float*  deg  = (float*)(ws + off);  off = alignup(off + (size_t)N * 4);
    __bf16* cW1s = (__bf16*)(ws + off); off = alignup(off + (size_t)LAYERS * TWO_D * D * 2);
    __bf16* cW2s = (__bf16*)(ws + off); off = alignup(off + (size_t)LAYERS * D * D * 2);
    float*  gmax = (float*)(ws + off);  off = alignup(off + (size_t)G * D * 4);
    float*  gsum = (float*)(ws + off);  off = alignup(off + (size_t)G * D * 4);
    float*  g2max= (float*)(ws + off);  off = alignup(off + (size_t)G * D * 4);
    float*  g2sum= (float*)(ws + off);  off = alignup(off + (size_t)G * D * 4);
    float*  gcnt = (float*)(ws + off);  off = alignup(off + (size_t)G * 4);

    // weight conversion + fragment shuffle to bf16 (per layer)
    for (int l = 0; l < LAYERS; ++l) {
        const int n1 = TWO_D * D;   // 8 col tiles x 8 k chunks
        const int n2 = D * D;       // 8 col tiles x 4 k chunks
        k_shuffleW<<<(n1 + 255) / 256, 256, 0, stream>>>(
            cW1 + (size_t)l * n1, cW1s + (size_t)l * n1, 8, n1);
        k_shuffleW<<<(n2 + 255) / 256, 256, 0, stream>>>(
            cW2 + (size_t)l * n2, cW2s + (size_t)l * n2, 4, n2);
    }
    // init graph buffers + degree
    k_fill_u32<<<(G * D + 255) / 256, 256, 0, stream>>>((unsigned int*)gmax,  NEG_INF_BITS, G * D);
    k_fill_u32<<<(G * D + 255) / 256, 256, 0, stream>>>((unsigned int*)g2max, NEG_INF_BITS, G * D);
    k_fill_u32<<<(G * D + 255) / 256, 256, 0, stream>>>((unsigned int*)gsum,  0u, G * D);
    k_fill_u32<<<(G * D + 255) / 256, 256, 0, stream>>>((unsigned int*)g2sum, 0u, G * D);
    k_fill_u32<<<(G + 255) / 256, 256, 0, stream>>>((unsigned int*)gcnt, 0u, G);
    k_fill_u32<<<(N + 255) / 256, 256, 0, stream>>>((unsigned int*)deg,  0u, N);

    // input net, input pooling, degrees
    k_input_net<<<N, 128, 0, stream>>>(x, in_W1, in_b1, in_g1, in_bn1,
                                       in_W2, in_b2, in_g2, in_bn2, xcur);
    k_seg_graph<<<N, 128, 0, stream>>>(xcur, batch, gmax, gsum, gcnt, 1);
    k_degree<<<(E + 255) / 256, 256, 0, stream>>>(dstI, deg, E);

    // edge-conv layers
    const int edgeBlocks = (E + EPB - 1) / EPB;
    for (int l = 0; l < LAYERS; ++l) {
        k_preln<<<N, 128, 0, stream>>>(xcur, pre_g + l * D, pre_b + l * D, hnb);
        k_init_aggr<<<((size_t)N * D + 255) / 256, 256, 0, stream>>>(amax, asum, N * D);
        k_edge_mlp<<<edgeBlocks, 128, 0, stream>>>(
            hnb, srcI, dstI,
            cW1s + (size_t)l * TWO_D * D, cb1 + l * D, cg1 + l * D, cbb1 + l * D,
            cW2s + (size_t)l * D * D,     cb2 + l * D, cg2 + l * D, cbb2 + l * D,
            amax, asum, E);
        k_node_update<<<N, 128, 0, stream>>>(amax, asum, deg,
                                             pW + (size_t)l * TWO_D * D, pb + l * D, xcur);
    }

    // final LN + pooling + head
    k_final_seg<<<N, 128, 0, stream>>>(xcur, ag_g, ag_b, batch, g2max, g2sum);
    k_graph_head<<<G, 128, 0, stream>>>(gmax, gsum, gcnt, g2max, g2sum,
                                        kW1, kb1, kg1, kbb1,
                                        kW2, kb2, kg2, kbb2,
                                        kW3, kb3, kg3, kbb3,
                                        kW4, kb4, (float*)d_out);
}
